// MultiheadGQA_26912265077095
// MI455X (gfx1250) — matrix operand order repaired and checked by reading, not compiled, not run on hardware
//
#include <hip/hip_runtime.h>
#include <hip/hip_bf16.h>
#include <math.h>

// ---------------------------------------------------------------------------
// GQA attention for MI455X (gfx1250, wave32, WMMA 16x16x32 f16 -> f32 accum)
//
// Shapes (fixed by reference): B=2, T=2048, C=2048, QH=32, KVH=8, D=64.
// Pipeline: cast(x,W*) -> f16 | Q/K/V GEMMs | V transpose | flash attention |
//           output GEMM (f32 out).
// WMMA fragment layouts (as used here; contiguous k, not the ISA table order):
//   A (16x32 f16): lane l, row = l&15, halves[0..15]=X[row][k0+16*(l>>4)+j]
//   B (32x16 f16): lane l, col = l&15, halves[0..15]=W[col][k0+16*(l>>4)+j]
//   C (16x16 f32): lane l, vgpr r -> C[r + 8*(l>>4)][l&15]
// ---------------------------------------------------------------------------

typedef __attribute__((ext_vector_type(16))) _Float16 v16h;
typedef __attribute__((ext_vector_type(8)))  _Float16 v8h;
typedef __attribute__((ext_vector_type(4)))  _Float16 v4h;
typedef __attribute__((ext_vector_type(8)))  float    v8f;

#define GQA_B   2
#define GQA_T   2048
#define GQA_C   2048
#define GQA_QH  32
#define GQA_KVH 8
#define GQA_D   64

static __device__ __forceinline__ v8f wmma_f16(v16h a, v16h b, v8f c) {
  return __builtin_amdgcn_wmma_f32_16x16x32_f16(
      /*neg_a=*/false, a, /*neg_b=*/false, b,
      /*c_mod=*/(short)0, c, /*reuse_a=*/false, /*reuse_b=*/false);
}

// A fragment from an f16 row (two aligned 16B loads).
static __device__ __forceinline__ v16h frag_a_f16(const _Float16* __restrict__ rowp,
                                                  int k0, int hl) {
  const v8h* p0 = (const v8h*)(rowp + k0 + 16 * hl);
  const v8h* p1 = (const v8h*)(rowp + k0 + 16 * hl + 8);
  v8h lo = *p0, hi = *p1;
  v16h a;
#pragma unroll
  for (int j = 0; j < 8; ++j) { a[j] = lo[j]; a[j + 8] = hi[j]; }
  return a;
}

// B fragment: 16 contiguous f16 at p (32B aligned).
static __device__ __forceinline__ v16h frag_b_f16(const _Float16* __restrict__ p) {
  const v8h* q = (const v8h*)p;
  v8h lo = q[0], hi = q[1];
  v16h b;
#pragma unroll
  for (int j = 0; j < 8; ++j) { b[j] = lo[j]; b[j + 8] = hi[j]; }
  return b;
}

// ---------------------------------------------------------------------------
// One-shot f32 -> f16 cast (n must be a multiple of 1024).
// ---------------------------------------------------------------------------
__global__ __launch_bounds__(256) void gqa_cast_f16(const float* __restrict__ src,
                                                    _Float16* __restrict__ dst,
                                                    int n4) {
  int idx = blockIdx.x * 256 + threadIdx.x;       // one float4 per thread
  if (idx >= n4) return;
  float4 v = ((const float4*)src)[idx];
  v4h o;
  o[0] = (_Float16)v.x; o[1] = (_Float16)v.y;
  o[2] = (_Float16)v.z; o[3] = (_Float16)v.w;
  ((v4h*)dst)[idx] = o;
}

// ---------------------------------------------------------------------------
// GEMM: O[M,N] (OutT) = X[M,K] (f16) @ W[N,K]^T (f16).
// Block = 256 threads = 8 waves; wave computes a 32(M) x 64(N) tile
// (2 A-fragments amortize each B-fragment: 8 WMMAs per k-step).
// Grid = (M/256, N/64).
// ---------------------------------------------------------------------------
template <typename OutT>
__global__ __launch_bounds__(256) void gqa_gemm_h(
    const _Float16* __restrict__ X, const _Float16* __restrict__ W,
    OutT* __restrict__ O, int M, int N, int K) {
  const int lane = threadIdx.x & 31;
  const int wave = threadIdx.x >> 5;
  const int hl = lane >> 4, l15 = lane & 15;
  const int m0 = blockIdx.x * 256 + wave * 32;
  const int n0 = blockIdx.y * 64;

  const _Float16* arow0 = X + (size_t)(m0 + l15) * K;
  const _Float16* arow1 = X + (size_t)(m0 + 16 + l15) * K;
  v8f acc[2][4] = {};
  for (int k0 = 0; k0 < K; k0 += 32) {
    v16h a0 = frag_a_f16(arow0, k0, hl);
    v16h a1 = frag_a_f16(arow1, k0, hl);
#pragma unroll
    for (int nt = 0; nt < 4; ++nt) {
      const _Float16* bp = W + (size_t)(n0 + nt * 16 + l15) * K + k0 + 16 * hl;
      v16h b = frag_b_f16(bp);
      acc[0][nt] = wmma_f16(a0, b, acc[0][nt]);
      acc[1][nt] = wmma_f16(a1, b, acc[1][nt]);
    }
  }
#pragma unroll
  for (int mt = 0; mt < 2; ++mt)
#pragma unroll
    for (int nt = 0; nt < 4; ++nt)
#pragma unroll
      for (int r = 0; r < 8; ++r) {
        int row = m0 + mt * 16 + r + 8 * hl;
        O[(size_t)row * N + n0 + nt * 16 + l15] = (OutT)acc[mt][nt][r];
      }
}

// ---------------------------------------------------------------------------
// Transpose V: Vh[B*T, KVH*D] (f16) -> Vt[B, KVH, D, T] (f16)
// ---------------------------------------------------------------------------
__global__ __launch_bounds__(256) void gqa_vtrans(const _Float16* __restrict__ V,
                                                  _Float16* __restrict__ Vt) {
  int idx = blockIdx.x * 256 + threadIdx.x;      // B*KVH*D*T = 2,097,152
  int t = idx & (GQA_T - 1);
  int d = (idx >> 11) & (GQA_D - 1);
  int g = (idx >> 17) & (GQA_KVH - 1);
  int b = idx >> 20;
  Vt[idx] = V[((size_t)b * GQA_T + t) * (GQA_KVH * GQA_D) + g * GQA_D + d];
}

// ---------------------------------------------------------------------------
// Causal GQA flash attention.
// Q : [B*T, QH*D] f16, Kh: [B*T, KVH*D] f16, Vt: [B, KVH, D, T] f16,
// Y : [B*T, QH*D] f16.
// Grid (T/128, QH, B); block 256 = 8 waves; each wave owns 16 query rows.
// ---------------------------------------------------------------------------
__global__ __launch_bounds__(256) void gqa_attn(
    const _Float16* __restrict__ Q, const _Float16* __restrict__ Kh,
    const _Float16* __restrict__ Vt, _Float16* __restrict__ Y) {
  __shared__ _Float16 pbuf[8][16][32];           // per-wave P staging (8 KB)

  const int lane = threadIdx.x & 31;
  const int wave = threadIdx.x >> 5;
  const int hl = lane >> 4, l15 = lane & 15;
  const int h = blockIdx.y;                      // query head 0..31
  const int b = blockIdx.z;
  const int g = h >> 2;                          // kv head = h / REPEATS
  const int qt0 = blockIdx.x * 128 + wave * 16;  // first query row of wave

  // Preload Q A-fragments for k-dim 64 (two k0 steps of 32).
  const _Float16* qrow =
      Q + ((size_t)b * GQA_T + qt0 + l15) * (GQA_QH * GQA_D) + h * GQA_D;
  const v16h qa0 = frag_a_f16(qrow, 0, hl);
  const v16h qa1 = frag_a_f16(qrow, 32, hl);

  v8f o[4] = {};
  float mrow[8], lrow[8];
#pragma unroll
  for (int r = 0; r < 8; ++r) { mrow[r] = -INFINITY; lrow[r] = 0.0f; }

  const float scale = 0.125f;                    // 1/sqrt(64)
  const int smax = qt0 + 15;                     // last valid key for this wave

  _Float16(*pb)[32] = pbuf[wave];

  for (int s0 = 0; s0 <= smax; s0 += 32) {
    // ---- scores: two 16-key tiles, k-dim 64 -> 4 WMMAs ----
    v8f s[2] = {};
#pragma unroll
    for (int st = 0; st < 2; ++st) {
      int srow = s0 + st * 16 + l15;
      const _Float16* kp =
          Kh + ((size_t)b * GQA_T + srow) * (GQA_KVH * GQA_D) + g * GQA_D + 16 * hl;
      s[st] = wmma_f16(qa0, frag_b_f16(kp), s[st]);
      s[st] = wmma_f16(qa1, frag_b_f16(kp + 32), s[st]);
    }

    // ---- scale, causal mask, online softmax update ----
#pragma unroll
    for (int r = 0; r < 8; ++r) {
      const int qr = qt0 + r + 8 * hl;
      float s0v = s[0][r] * scale;
      float s1v = s[1][r] * scale;
      if (s0 + l15 > qr)      s0v = -INFINITY;
      if (s0 + 16 + l15 > qr) s1v = -INFINITY;
      float mx = fmaxf(s0v, s1v);
#pragma unroll
      for (int off = 1; off < 16; off <<= 1)
        mx = fmaxf(mx, __shfl_xor(mx, off, 32));  // within 16-lane half
      const float mnew = fmaxf(mrow[r], mx);
      const float p0 = __expf(s0v - mnew);
      const float p1 = __expf(s1v - mnew);
      s[0][r] = p0; s[1][r] = p1;
      float sum = p0 + p1;
#pragma unroll
      for (int off = 1; off < 16; off <<= 1)
        sum += __shfl_xor(sum, off, 32);
      const float corr = __expf(mrow[r] - mnew);
      lrow[r] = lrow[r] * corr + sum;
      mrow[r] = mnew;
#pragma unroll
      for (int nt = 0; nt < 4; ++nt) o[nt][r] *= corr;
    }

    // ---- P: C-layout -> LDS -> A-layout ----
#pragma unroll
    for (int r = 0; r < 8; ++r) {
      const int pr = r + 8 * hl;
      pb[pr][l15]      = (_Float16)s[0][r];
      pb[pr][16 + l15] = (_Float16)s[1][r];
    }
    asm volatile("s_wait_dscnt 0" ::: "memory");  // wave-local LDS RAW fence
    v16h pa;
    {
      const v8h* p0 = (const v8h*)&pb[l15][16 * hl];
      const v8h* p1 = (const v8h*)&pb[l15][16 * hl + 8];
      v8h lo = *p0, hi = *p1;
#pragma unroll
      for (int j = 0; j < 8; ++j) { pa[j] = lo[j]; pa[j + 8] = hi[j]; }
    }

    // ---- O += P @ V : 4 WMMAs over d tiles ----
#pragma unroll
    for (int nt = 0; nt < 4; ++nt) {
      const int d = nt * 16 + l15;
      const _Float16* vp =
          Vt + (((size_t)(b * GQA_KVH + g) * GQA_D + d) * GQA_T) + s0 + 16 * hl;
      o[nt] = wmma_f16(pa, frag_b_f16(vp), o[nt]);
    }
  }

  // ---- normalize and store ----
#pragma unroll
  for (int nt = 0; nt < 4; ++nt)
#pragma unroll
    for (int r = 0; r < 8; ++r) {
      const int qr = qt0 + r + 8 * hl;
      const float val = o[nt][r] / lrow[r];
      Y[((size_t)b * GQA_T + qr) * (GQA_QH * GQA_D) + h * GQA_D + nt * 16 + l15] =
          (_Float16)val;
    }
}

// ---------------------------------------------------------------------------
// Launch: casts -> Q/K/V projections -> V transpose -> attention -> out proj.
// ---------------------------------------------------------------------------
extern "C" void kernel_launch(void* const* d_in, const int* in_sizes, int n_in,
                              void* d_out, int out_size, void* d_ws, size_t ws_size,
                              hipStream_t stream) {
  (void)in_sizes; (void)n_in; (void)out_size; (void)ws_size;
  const float* x  = (const float*)d_in[0];
  const float* Wq = (const float*)d_in[1];
  const float* Wk = (const float*)d_in[2];
  const float* Wv = (const float*)d_in[3];
  const float* Wo = (const float*)d_in[4];
  float* out = (float*)d_out;

  const int BT = GQA_B * GQA_T;                     // 4096
  const size_t nX  = (size_t)BT * GQA_C;            // 8,388,608
  const size_t nWq = (size_t)GQA_C * GQA_C;         // 4,194,304
  const size_t nWk = (size_t)(GQA_KVH * GQA_D) * GQA_C;  // 1,048,576
  const size_t nQ  = (size_t)BT * GQA_QH * GQA_D;   // 8,388,608
  const size_t nKV = (size_t)BT * GQA_KVH * GQA_D;  // 2,097,152

  _Float16* xh  = (_Float16*)d_ws;
  _Float16* wqh = xh + nX;
  _Float16* wkh = wqh + nWq;
  _Float16* wvh = wkh + nWk;
  _Float16* woh = wvh + nWk;
  _Float16* qh  = woh + nWq;
  _Float16* kh  = qh + nQ;
  _Float16* vh  = kh + nKV;
  _Float16* vt  = vh + nKV;
  _Float16* yh  = vt + nKV;                         // total ~84 MB of ws

  dim3 blk(256);
  // ---- one-shot f32 -> f16 casts ----
  gqa_cast_f16<<<(int)(nX / 1024),  blk, 0, stream>>>(x,  xh,  (int)(nX / 4));
  gqa_cast_f16<<<(int)(nWq / 1024), blk, 0, stream>>>(Wq, wqh, (int)(nWq / 4));
  gqa_cast_f16<<<(int)(nWk / 1024), blk, 0, stream>>>(Wk, wkh, (int)(nWk / 4));
  gqa_cast_f16<<<(int)(nWk / 1024), blk, 0, stream>>>(Wv, wvh, (int)(nWk / 4));
  gqa_cast_f16<<<(int)(nWq / 1024), blk, 0, stream>>>(Wo, woh, (int)(nWq / 4));

  // ---- projections (all-f16 WMMA GEMMs) ----
  gqa_gemm_h<_Float16><<<dim3(BT / 256, GQA_C / 64), blk, 0, stream>>>(
      xh, wqh, qh, BT, GQA_QH * GQA_D, GQA_C);
  gqa_gemm_h<_Float16><<<dim3(BT / 256, (GQA_KVH * GQA_D) / 64), blk, 0, stream>>>(
      xh, wkh, kh, BT, GQA_KVH * GQA_D, GQA_C);
  gqa_gemm_h<_Float16><<<dim3(BT / 256, (GQA_KVH * GQA_D) / 64), blk, 0, stream>>>(
      xh, wvh, vh, BT, GQA_KVH * GQA_D, GQA_C);

  // ---- V transpose to [B, KVH, D, T] ----
  gqa_vtrans<<<(int)(nKV / 256), blk, 0, stream>>>(vh, vt);

  // ---- flash attention ----
  gqa_attn<<<dim3(GQA_T / 128, GQA_QH, GQA_B), blk, 0, stream>>>(qh, kh, vt, yh);

  // ---- output projection: f32 result ----
  gqa_gemm_h<float><<<dim3(BT / 256, GQA_C / 64), blk, 0, stream>>>(
      yh, woh, out, BT, GQA_C, GQA_QH * GQA_D);
}
